// SelfAttention_Head_43190191129018
// MI455X (gfx1250) — matrix operand order repaired and checked by reading, not compiled, not run on hardware
//
#include <hip/hip_runtime.h>

#define DEVINL __device__ __forceinline__

typedef __attribute__((ext_vector_type(16))) _Float16 v16h;
typedef __attribute__((ext_vector_type(8)))  _Float16 v8h;
typedef __attribute__((ext_vector_type(8)))  float    v8f;
typedef __attribute__((ext_vector_type(4)))  float    v4f;
typedef __attribute__((ext_vector_type(4)))  int      v4i;

constexpr int Bn = 8, Tt = 4096, Cc = 1024, Dd = 128;
constexpr int BT = Bn * Tt;
constexpr float SM_SCALE = 0.08838834764831845f; // 1/sqrt(128)

DEVINL v16h cat16(v8h lo, v8h hi) {
  return __builtin_shufflevector(lo, hi, 0,1,2,3,4,5,6,7,8,9,10,11,12,13,14,15);
}

// Dense B-fragment (32x16, f16): lane holds N=l&15, elements K=(l>>4)*16+i -> 16 contiguous f16.
DEVINL v16h load_b_f16(const _Float16* p) {
  return cat16(*(const v8h*)p, *(const v8h*)(p + 8));
}

DEVINL v8f wmma16(v16h a, v16h b, v8f c) {
  return __builtin_amdgcn_wmma_f32_16x16x32_f16(false, a, false, b, (short)0, c, false, false);
}

DEVINL void zero8(v8f a[8]) {
  #pragma unroll
  for (int n = 0; n < 8; ++n)
    #pragma unroll
    for (int i = 0; i < 8; ++i) a[n][i] = 0.0f;
}

// --- CDNA5 async memory->LDS copy (16B per lane), tracked by ASYNCcnt -------
DEVINL void async_copy16(_Float16* lds_dst, const _Float16* gsrc) {
#if __has_builtin(__builtin_amdgcn_global_load_async_to_lds_b128)
  // builtin signature (from hipcc diagnostic): (AS1 v4i*, AS3 v4i*, int, int)
  void* g = (void*)gsrc;   // strip const on generic pointer first
  void* l = (void*)lds_dst;
  __builtin_amdgcn_global_load_async_to_lds_b128(
      (__attribute__((address_space(1))) v4i*)g,
      (__attribute__((address_space(3))) v4i*)l, 0, 0);
#else
  // GV mode: LDS address from the vdst VGPR, global address from 64-bit vaddr
  unsigned lds_off = (unsigned)(unsigned long long)lds_dst; // low 32b of generic = LDS offset
  asm volatile("global_load_async_to_lds_b128 %0, %1, off"
               :: "v"(lds_off), "v"(gsrc) : "memory");
#endif
}

DEVINL void wait_async0() {
#if __has_builtin(__builtin_amdgcn_s_wait_asynccnt)
  __builtin_amdgcn_s_wait_asynccnt(0);
#else
  asm volatile("s_wait_asynccnt 0x0" ::: "memory");
#endif
}

// ---------------------------------------------------------------------------
// Kernel 0: transpose+convert weights: wt_w[d*C + c] = (f16) w[c*D + d], w in {q,k,v}
// ---------------------------------------------------------------------------
__global__ void wconv_kernel(const float* __restrict__ wq, const float* __restrict__ wk,
                             const float* __restrict__ wv, _Float16* __restrict__ wt) {
  int idx = blockIdx.x * 256 + threadIdx.x;
  const int per = Cc * Dd;
  if (idx >= 3 * per) return;
  int w = idx / per, rem = idx - w * per;
  int d = rem / Cc, c = rem - d * Cc;
  const float* s = (w == 0) ? wq : (w == 1) ? wk : wv;
  wt[idx] = (_Float16)s[(size_t)c * Dd + d];
}

// ---------------------------------------------------------------------------
// Per-wave 16x128 GEMM tile over C=1024 (A from fp32 x, B from f16 transposed W)
// B-fragments are preloaded per k-chunk so the 16 loads clause together.
// ---------------------------------------------------------------------------
DEVINL void gemm_1024(const float* __restrict__ xp, const _Float16* __restrict__ wt_o,
                      int koff, int col, int kg, v8f acc[8]) {
  for (int cc = 0; cc < 32; ++cc) {
    const int cb = cc * 32;
    // A fragment 16x32 f16: lane row = l&15 (xp already offset), elements:
    // i -> K=cb+kg+i (contiguous)
    v4f a0 = *(const v4f*)(xp + cb + kg);
    v4f a1 = *(const v4f*)(xp + cb + kg + 4);
    v4f a2 = *(const v4f*)(xp + cb + kg + 8);
    v4f a3 = *(const v4f*)(xp + cb + kg + 12);
    v8h lo, hi;
    #pragma unroll
    for (int i = 0; i < 4; ++i) {
      lo[i] = (_Float16)a0[i]; lo[i + 4] = (_Float16)a1[i];
      hi[i] = (_Float16)a2[i]; hi[i + 4] = (_Float16)a3[i];
    }
    v16h a = cat16(lo, hi);
    v16h bfr[8];
    #pragma unroll
    for (int nt = 0; nt < 8; ++nt)
      bfr[nt] = load_b_f16(wt_o + (size_t)(nt * 16 + col) * Cc + cb + kg);
    #pragma unroll
    for (int nt = 0; nt < 8; ++nt)
      acc[nt] = wmma16(a, bfr[nt], acc[nt]);
  }
}

// ---------------------------------------------------------------------------
// Kernel 1: q,k,v projections. Block = 128 thr (4 waves), wave owns 16 rows.
// q16,k16: f16 [B*T][D]; vT: f16 [B][D][T] (transposed via per-wave LDS bounce)
// ---------------------------------------------------------------------------
__global__ __launch_bounds__(128) void proj_qkv_kernel(
    const float* __restrict__ x, const _Float16* __restrict__ wt,
    _Float16* __restrict__ q16, _Float16* __restrict__ k16, _Float16* __restrict__ vT) {
  __shared__ _Float16 vs[4][Dd * 16];
  const int tid = threadIdx.x;
  const int wave = tid >> 5, lane = tid & 31;
  const int row0 = blockIdx.x * 64 + wave * 16;        // global row in [0, B*T)
  const int col = lane & 15;
  const int koff = (lane >> 4) * 8;
  const int kg   = (lane >> 4) * 16;
  const int m0   = (lane >> 4) * 8;
  const float* xp = x + (size_t)(row0 + col) * Cc;     // A-frag row = lane&15

  v8f acc[8];

  // ---- Q (pre-scaled by 1/sqrt(D)) ----
  zero8(acc);
  gemm_1024(xp, wt, koff, col, kg, acc);
  {
    _Float16* dq = q16 + (size_t)row0 * Dd;
    #pragma unroll
    for (int nt = 0; nt < 8; ++nt)
      #pragma unroll
      for (int i = 0; i < 8; ++i)
        dq[(size_t)(m0 + i) * Dd + nt * 16 + col] = (_Float16)(acc[nt][i] * SM_SCALE);
  }

  // ---- K ----
  zero8(acc);
  gemm_1024(xp, wt + (size_t)Dd * Cc, koff, col, kg, acc);
  {
    _Float16* dk = k16 + (size_t)row0 * Dd;
    #pragma unroll
    for (int nt = 0; nt < 8; ++nt)
      #pragma unroll
      for (int i = 0; i < 8; ++i)
        dk[(size_t)(m0 + i) * Dd + nt * 16 + col] = (_Float16)acc[nt][i];
  }

  // ---- V (store transposed [B][D][T]) ----
  zero8(acc);
  gemm_1024(xp, wt + (size_t)2 * Dd * Cc, koff, col, kg, acc);
  {
    _Float16* myvs = vs[wave];                     // [d][t_local] = [128][16]
    #pragma unroll
    for (int nt = 0; nt < 8; ++nt)
      #pragma unroll
      for (int i = 0; i < 8; ++i)
        myvs[(nt * 16 + col) * 16 + (m0 + i)] = (_Float16)acc[nt][i];
    // same-wave DS ops are in-order: safe to read back without a barrier
    const int bb = row0 / Tt, t0 = row0 % Tt;
    _Float16* vb = vT + (size_t)bb * Dd * Tt;
    #pragma unroll
    for (int rr = 0; rr < 4; ++rr) {
      const int d = rr * 32 + lane;
      const _Float16* srow = myvs + d * 16;
      _Float16* drow = vb + (size_t)d * Tt + t0;
      *(v8h*)drow       = *(const v8h*)srow;
      *(v8h*)(drow + 8) = *(const v8h*)(srow + 8);
    }
  }
}

// ---------------------------------------------------------------------------
// Kernel 2: causal flash attention. Block = 128 thr (4 waves) owns 64 q-rows,
// each wave owns 16 q-rows x D=128. K/V 32-key tiles double-buffered in LDS,
// filled with GLOBAL_LOAD_ASYNC_TO_LDS_B128 (ASYNCcnt) overlapping compute.
// ---------------------------------------------------------------------------
__global__ __launch_bounds__(128) void attn_kernel(
    const _Float16* __restrict__ q16, const _Float16* __restrict__ k16,
    const _Float16* __restrict__ vT, float* __restrict__ out) {
  __shared__ _Float16 Ks[2][32 * Dd];   // [buf][key][d]
  __shared__ _Float16 Vs[2][Dd * 32];   // [buf][d][key]
  __shared__ _Float16 Ps[4][16 * 32];   // per-wave P relayout scratch

  const int tid = threadIdx.x;
  const int wave = tid >> 5, lane = tid & 31;
  const int b = blockIdx.y;
  const int qb0 = blockIdx.x * 64;
  const int q_wb = qb0 + wave * 16;
  const int col = lane & 15;
  const int koff = (lane >> 4) * 8;
  const int kg   = (lane >> 4) * 16;
  const int m0   = (lane >> 4) * 8;

  const _Float16* kb_base = k16 + (size_t)b * Tt * Dd;
  const _Float16* vb_base = vT + (size_t)b * Dd * Tt;

  // async-stage one 32-key K/V tile into LDS buffer `bufi` (all lanes active)
  auto stage = [&](int kb, int bufi) {
    const int kbase = kb << 5;
    _Float16* KsB = Ks[bufi];
    _Float16* VsB = Vs[bufi];
    #pragma unroll
    for (int j = 0; j < 4; ++j) {
      const int chunk = j * 128 + tid;                 // 0..511, 16B each
      const int key = chunk >> 4, dq = (chunk & 15) * 8;
      async_copy16(KsB + key * Dd + dq,
                   kb_base + (size_t)(kbase + key) * Dd + dq);
      const int d = chunk >> 2, kq = (chunk & 3) * 8;
      async_copy16(VsB + d * 32 + kq,
                   vb_base + (size_t)d * Tt + kbase + kq);
    }
  };

  // Q fragments: 4 chunks of K-dim (d), loop invariant over key blocks
  v16h qf[4];
  {
    const _Float16* qp = q16 + (size_t)(b * Tt + q_wb + col) * Dd;
    #pragma unroll
    for (int cc = 0; cc < 4; ++cc) {
      const _Float16* p = qp + cc * 32 + kg;
      qf[cc] = cat16(*(const v8h*)p, *(const v8h*)(p + 8));
    }
  }

  v8f acc[8];
  zero8(acc);
  v8f mrow, lrow;
  #pragma unroll
  for (int i = 0; i < 8; ++i) { mrow[i] = -__builtin_inff(); lrow[i] = 0.0f; }

  const int nblk = (qb0 + 64) >> 5;          // key blocks needed by whole block
  const int wlim = (q_wb + 16 + 31) >> 5;    // key blocks needed by this wave

  stage(0, 0);                               // prologue: prefetch first tile

  for (int kb = 0; kb < nblk; ++kb) {
    const int buf = kb & 1;
    wait_async0();                           // my async stores to LDS landed
    __syncthreads();                         // everyone's landed; prev reads done
    if (kb + 1 < nblk) stage(kb + 1, buf ^ 1);  // overlap next tile with compute
    if (kb >= wlim) continue;                // beyond causal bound for this wave

    const int kbase = kb << 5;
    const _Float16* KsB = Ks[buf];
    const _Float16* VsB = Vs[buf];

    // S = Q K^T  (16 x 32 keys), accumulated over 4 d-chunks
    v8f s0, s1;
    #pragma unroll
    for (int i = 0; i < 8; ++i) { s0[i] = 0.0f; s1[i] = 0.0f; }
    #pragma unroll
    for (int cc = 0; cc < 4; ++cc) {
      const _Float16* p0 = KsB + (size_t)col * Dd + cc * 32 + kg;
      const _Float16* p1 = KsB + (size_t)(16 + col) * Dd + cc * 32 + kg;
      s0 = wmma16(qf[cc], load_b_f16(p0), s0);
      s1 = wmma16(qf[cc], load_b_f16(p1), s1);
    }

    // causal mask (diagonal blocks only); Q was pre-scaled in projection
    if (kbase + 31 > q_wb) {
      #pragma unroll
      for (int i = 0; i < 8; ++i) {
        const int q = q_wb + m0 + i;
        if (kbase + col > q)      s0[i] = -__builtin_inff();
        if (kbase + 16 + col > q) s1[i] = -__builtin_inff();
      }
    }

    // online softmax: row max over 32 keys (butterfly over 16 lanes of half-wave)
    v8f mb;
    #pragma unroll
    for (int i = 0; i < 8; ++i) mb[i] = fmaxf(s0[i], s1[i]);
    #pragma unroll
    for (int msk = 1; msk <= 8; msk <<= 1)
      #pragma unroll
      for (int i = 0; i < 8; ++i)
        mb[i] = fmaxf(mb[i], __shfl_xor(mb[i], msk, 32));

    v8f alpha;
    #pragma unroll
    for (int i = 0; i < 8; ++i) {
      const float mn = fmaxf(mrow[i], mb[i]);
      alpha[i] = __expf(mrow[i] - mn);
      mrow[i] = mn;
      s0[i] = __expf(s0[i] - mn);
      s1[i] = __expf(s1[i] - mn);
    }
    v8f rs;
    #pragma unroll
    for (int i = 0; i < 8; ++i) rs[i] = s0[i] + s1[i];
    #pragma unroll
    for (int msk = 1; msk <= 8; msk <<= 1)
      #pragma unroll
      for (int i = 0; i < 8; ++i)
        rs[i] += __shfl_xor(rs[i], msk, 32);
    #pragma unroll
    for (int i = 0; i < 8; ++i) lrow[i] = lrow[i] * alpha[i] + rs[i];
    #pragma unroll
    for (int nt = 0; nt < 8; ++nt)
      #pragma unroll
      for (int i = 0; i < 8; ++i) acc[nt][i] *= alpha[i];

    // P (f16) -> per-wave LDS scratch, re-read in A-fragment layout
    _Float16* ps = Ps[wave];
    #pragma unroll
    for (int i = 0; i < 8; ++i) {
      ps[(m0 + i) * 32 + col]      = (_Float16)s0[i];
      ps[(m0 + i) * 32 + col + 16] = (_Float16)s1[i];
    }
    const _Float16* pp = ps + col * 32 + kg;   // same-wave DS in-order
    v16h pf = cat16(*(const v8h*)pp, *(const v8h*)(pp + 8));

    // O += P V   (8 d-tiles of 16)
    #pragma unroll
    for (int nt = 0; nt < 8; ++nt) {
      const _Float16* vp = VsB + (size_t)(nt * 16 + col) * 32 + kg;
      acc[nt] = wmma16(pf, load_b_f16(vp), acc[nt]);
    }
  }

  // epilogue: normalize and store fp32
  v8f inv;
  #pragma unroll
  for (int i = 0; i < 8; ++i) inv[i] = 1.0f / lrow[i];
  float* op = out + (size_t)(b * Tt + q_wb) * Dd;
  #pragma unroll
  for (int nt = 0; nt < 8; ++nt)
    #pragma unroll
    for (int i = 0; i < 8; ++i)
      op[(size_t)(m0 + i) * Dd + nt * 16 + col] = acc[nt][i] * inv[i];
}

// ---------------------------------------------------------------------------
extern "C" void kernel_launch(void* const* d_in, const int* in_sizes, int n_in,
                              void* d_out, int out_size, void* d_ws, size_t ws_size,
                              hipStream_t stream) {
  (void)in_sizes; (void)n_in; (void)out_size; (void)ws_size;
  const float* x  = (const float*)d_in[0];
  const float* wq = (const float*)d_in[1];
  const float* wk = (const float*)d_in[2];
  const float* wv = (const float*)d_in[3];
  float* out = (float*)d_out;

  // workspace layout (f16): q16[BT*D] | k16[BT*D] | vT[B*D*T] | wt[3*D*C]
  _Float16* q16 = (_Float16*)d_ws;
  _Float16* k16 = q16 + (size_t)BT * Dd;
  _Float16* vT  = k16 + (size_t)BT * Dd;
  _Float16* wt  = vT  + (size_t)BT * Dd;

  wconv_kernel<<<(3 * Cc * Dd + 255) / 256, 256, 0, stream>>>(wq, wk, wv, wt);
  proj_qkv_kernel<<<BT / 64, 128, 0, stream>>>(x, wt, q16, k16, vT);
  attn_kernel<<<dim3(Tt / 64, Bn), 128, 0, stream>>>(q16, k16, vT, out);
}
